// SrIEM_37692632990373
// MI455X (gfx1250) — matrix operand order repaired and checked by reading, not compiled, not run on hardware
//
#include <hip/hip_runtime.h>
#include <hip/hip_bf16.h>

#define B_  1024
#define L_  50
#define H_  256
#define N_  100000
#define A_  100
#define NC_ 99999   // N-1

typedef __attribute__((ext_vector_type(16))) __bf16 v16bf;
typedef __attribute__((ext_vector_type(8)))  __bf16 v8bf;
typedef __attribute__((ext_vector_type(4)))  __bf16 v4bf;
typedef __attribute__((ext_vector_type(8)))  float  v8f;
typedef __attribute__((ext_vector_type(4)))  float  v4f;

__device__ __forceinline__ v8f wmma_bf(v16bf a, v16bf b, v8f c) {
    // D = A(16x32 bf16) * B(32x16 bf16) + C(16x16 f32)
    return __builtin_amdgcn_wmma_f32_16x16x32_bf16(
        false, a, false, b, (short)0, c, false, false);
}

__device__ __forceinline__ v16bf pack2(v8bf lo, v8bf hi) {
    union { v16bf v; v8bf h[2]; } u;
    u.h[0] = lo; u.h[1] = hi;
    return u.v;
}

// load 16 contiguous f32 (64B-aligned) and convert to bf16 vector
__device__ __forceinline__ v16bf cvt16(const float* __restrict__ p) {
    const v4f* q = (const v4f*)p;
    v16bf r;
#pragma unroll
    for (int i = 0; i < 4; ++i) {
        v4f f = q[i];
        r[4*i+0] = (__bf16)f.x; r[4*i+1] = (__bf16)f.y;
        r[4*i+2] = (__bf16)f.z; r[4*i+3] = (__bf16)f.w;
    }
    return r;
}

// fast sigmoid: result is rounded to bf16 afterwards, so v_rcp_f32 is plenty
__device__ __forceinline__ float fast_sigmoid(float x) {
    return __builtin_amdgcn_rcpf(1.0f + __expf(-x));
}

// ---- session-kernel LDS layout (bytes) ----
#define XF_OFF    0          // f32  [50][260]  = 52000 B
#define XBF_OFF   52000      // bf16 [64][264]  = 33792 B
#define QBF_OFF   85792      // bf16 [64][136]  = 17408 B
#define KBF_OFF   103200     // bf16 [64][136]  = 17408 B
#define AFF_OFF   120608     // f32  [64][68]   = 17408 B
#define MASK_OFF  138016     // f32  [64]
#define P_OFF     138272     // f32  [64]
#define SC_OFF    138528     // f32  [64]
#define ATT_OFF   138784     // f32  [256]
#define LAST_OFF  139808     // f32  [256]
#define LEN_OFF   140832     // int
#define SMEM_TOTAL 140848

__global__ __launch_bounds__(256)
void sriem_session_kernel(const int* __restrict__ inputs,
                          const float* __restrict__ masks,
                          const float* __restrict__ emb,
                          const float* __restrict__ Qw,
                          const float* __restrict__ Kw,
                          const float* __restrict__ lin_w,
                          const float* __restrict__ lin_b,
                          float* __restrict__ out_aff,
                          __bf16* __restrict__ wsA)
{
    extern __shared__ char smem[];
    float*  xf     = (float*)(smem + XF_OFF);     // row stride 260 f32
    __bf16* xbf    = (__bf16*)(smem + XBF_OFF);   // row stride 264
    __bf16* Qbf    = (__bf16*)(smem + QBF_OFF);   // row stride 136
    __bf16* Kbf    = (__bf16*)(smem + KBF_OFF);   // row stride 136
    float*  aff    = (float*)(smem + AFF_OFF);    // row stride 68 f32
    float*  mask_s = (float*)(smem + MASK_OFF);
    float*  p_s    = (float*)(smem + P_OFF);
    float*  sc_s   = (float*)(smem + SC_OFF);
    float*  att_s  = (float*)(smem + ATT_OFF);
    float*  last_s = (float*)(smem + LAST_OFF);
    int*    len_s  = (int*)(smem + LEN_OFF);

    const int tid  = threadIdx.x;
    const int b    = blockIdx.x;
    const int lane = tid & 31;
    const int wave = tid >> 5;
    const int nl   = lane & 15;
    const int kh   = lane >> 4;

    // zero the bf16 matrices (padding must be true zeros) + scratch
    {
        uint32_t* z = (uint32_t*)(smem + XBF_OFF);
        for (int i = tid; i < (SMEM_TOTAL - XBF_OFF) / 4; i += 256) z[i] = 0u;
    }
    __syncthreads();

    // ---- gather x = emb[inputs[b]] into LDS (f32 + bf16) ----
    for (int l = 0; l < L_; ++l) {
        int idx = inputs[b * L_ + l];
        float v = emb[(size_t)idx * H_ + tid];
        xf[l * 260 + tid]  = v;
        xbf[l * 264 + tid] = (__bf16)v;
    }
    if (tid < L_) mask_s[tid] = masks[b * L_ + tid];
    __syncthreads();

    // ---- Q,K = sigmoid(x @ {Qw,Kw}^T) via WMMA bf16 ----
    // M=64 (4 tiles, rows>=50 are zero pads), N=112 (7 tiles, a>=100 zero-selected), K=256
    for (int t = wave * 7; t < wave * 7 + 7; ++t) {
        int qk = t / 28, rem = t % 28;
        int mt = rem / 7, at = rem % 7;
        const float* W = qk ? Kw : Qw;
        __bf16* dst    = qk ? Kbf : Qbf;
        const int acol = at * 16 + nl;
        const int ac   = (acol < A_) ? acol : 0;   // clamp, select below (no exec branch)
        const bool ok  = (acol < A_);
        v8f c = 0;
#pragma unroll 2
        for (int ks = 0; ks < 8; ++ks) {
            int kb  = ks * 32;
            int row = mt * 16 + nl;  // A: M = lane&15
            v8bf lo = *(const v8bf*)(xbf + row * 264 + kb + kh * 16);
            v8bf hi = *(const v8bf*)(xbf + row * 264 + kb + kh * 16 + 8);
            v16bf a = pack2(lo, hi);
            v16bf z{};
            v16bf t16 = cvt16(W + ac * H_ + kb + kh * 16);
            v16bf bm  = ok ? t16 : z;
            c = wmma_bf(a, bm, c);
        }
#pragma unroll
        for (int r = 0; r < 8; ++r) {
            int row = mt * 16 + r + (kh << 3);
            int col = at * 16 + nl;
            if (col < A_)
                dst[row * 136 + col] = (__bf16)fast_sigmoid(c[r]);
        }
    }
    __syncthreads();

    // ---- affinity = Q @ K^T / 10 : 4x4 tiles, K padded 100->128 (zeros) ----
#pragma unroll
    for (int tt = 0; tt < 2; ++tt) {
        int t  = wave * 2 + tt;
        int mt = t >> 2, nt = t & 3;
        v8f c = 0;
#pragma unroll
        for (int ks = 0; ks < 4; ++ks) {
            int kb   = ks * 32;
            int arow = mt * 16 + nl;
            v8bf lo  = *(const v8bf*)(Qbf + arow * 136 + kb + kh * 16);
            v8bf hi  = *(const v8bf*)(Qbf + arow * 136 + kb + kh * 16 + 8);
            v16bf a  = pack2(lo, hi);
            int brow = nt * 16 + nl;   // B column n = K row n (contiguous)
            v8bf blo = *(const v8bf*)(Kbf + brow * 136 + kb + kh * 16);
            v8bf bhi = *(const v8bf*)(Kbf + brow * 136 + kb + kh * 16 + 8);
            c = wmma_bf(a, pack2(blo, bhi), c);
        }
#pragma unroll
        for (int r = 0; r < 8; ++r) {
            int row = mt * 16 + r + (kh << 3);
            int col = nt * 16 + nl;
            float v = c[r] * 0.1f;   // 1/sqrt(A), A=100
            aff[row * 68 + col] = v;
            if (row < L_ && col < L_)
                out_aff[(size_t)b * (L_ * L_) + row * L_ + col] = v;
        }
    }
    __syncthreads();

    // ---- item score: masked row sums, remove diagonal ----
    if (tid < L_) {
        float s = 0.0f;
        for (int m = 0; m < L_; ++m) s += aff[tid * 68 + m] * mask_s[m];
        s -= aff[tid * 68 + tid] * mask_s[tid];
        sc_s[tid] = s * mask_s[tid];
    }
    __syncthreads();
    if (tid == 0) {
        float mx = -3.4e38f;
        for (int l = 0; l < L_; ++l) mx = fmaxf(mx, sc_s[l]);
        float sum = 0.0f;
        for (int l = 0; l < L_; ++l) { float e = __expf(sc_s[l] - mx); p_s[l] = e; sum += e; }
        float isum = 1.0f / sum;
        float s2 = 0.0f;
        for (int l = 0; l < L_; ++l) { float p = p_s[l] * isum * mask_s[l]; p_s[l] = p; s2 += p; }
        float inv = 1.0f / s2;
        for (int l = 0; l < L_; ++l) p_s[l] *= inv;
        float ts = 0.0f;
        for (int l = 0; l < L_; ++l) ts += mask_s[l];
        *len_s = (int)ts;
    }
    __syncthreads();

    // ---- attention pooling + last-item embedding ----
    {
        float att = 0.0f;
        for (int l = 0; l < L_; ++l) att += p_s[l] * xf[l * 260 + tid];
        att_s[tid] = att;
        int lid = inputs[b * L_ + (*len_s - 1)];
        last_s[tid] = emb[(size_t)lid * H_ + tid];
    }
    __syncthreads();

    // ---- final_rep = concat(att,last) @ lin_w^T + lin_b; store WMMA-A-swizzled bf16 ----
    {
        float acc = lin_b[tid];
        const float* wr = lin_w + tid * (2 * H_);
        for (int j = 0; j < H_; j += 4)
            acc += att_s[j] * wr[j] + att_s[j+1] * wr[j+1]
                 + att_s[j+2] * wr[j+2] + att_s[j+3] * wr[j+3];
        const float* wr2 = wr + H_;
        for (int j = 0; j < H_; j += 4)
            acc += last_s[j] * wr2[j] + last_s[j+1] * wr2[j+1]
                 + last_s[j+2] * wr2[j+2] + last_s[j+3] * wr2[j+3];
        // 16-bit A(16x32) lane layout: lane = m + 16*((k%16)/8), elem = k%8 + 8*(k/16)
        int m = b & 15, mtile = b >> 4;
        int ksn = tid >> 5, kk = tid & 31;
        int lane2 = m + 16 * ((kk & 15) >> 3);
        int elem  = (kk & 7) + 8 * (kk >> 4);
        wsA[(size_t)(mtile * 8 + ksn) * 512 + lane2 * 16 + elem] = (__bf16)acc;
    }
}

// scores[m,n] = sum_k final_rep[m,k] * emb[1+n,k]
// workgroup tile: 256(M, via blockIdx.y) x 64(N).
// B panel (64 emb rows x 256 K, bf16) is staged ONCE per workgroup into LDS;
// the WMMA loop runs guard-free (edge columns are zeros in LDS).
#define BS_STRIDE 264   // bf16 elems per row (8-elem pad rotates banks)

__global__ __launch_bounds__(256)
void sriem_scores_kernel(const __bf16* __restrict__ wsA,
                         const float* __restrict__ emb,
                         float* __restrict__ out)
{
    __shared__ __bf16 Bs[64 * BS_STRIDE];   // 33 KB

    const int tid  = threadIdx.x;
    const int lane = tid & 31;
    const int wave = tid >> 5;
    const int nl   = lane & 15;
    const int kh   = lane >> 4;
    const int mt0  = blockIdx.y * 16 + wave * 2;
    const int mt1  = mt0 + 1;
    const int nbase = blockIdx.x * 64;
    const bool full = (nbase + 64 <= NC_);

    // ---- cooperative stage: emb[nbase+1 .. nbase+64] -> bf16 LDS panel ----
    {
        const int r    = (tid >> 6);        // +4 per iter, 0..63 total
        const int cseg = (tid & 63) * 4;    // 0..252
#pragma unroll 4
        for (int rr = 0; rr < 16; ++rr) {
            int row = rr * 4 + r;           // 0..63  (B column = emb row nbase+row+1)
            int col = nbase + row;
            v4f f = {0.0f, 0.0f, 0.0f, 0.0f};
            if (col < NC_) f = *(const v4f*)(emb + (size_t)(col + 1) * H_ + cseg);
            v4bf pk = { (__bf16)f.x, (__bf16)f.y, (__bf16)f.z, (__bf16)f.w };
            *(v4bf*)(Bs + row * BS_STRIDE + cseg) = pk;
        }
    }
    __syncthreads();

    v8f c[8];
#pragma unroll
    for (int i = 0; i < 8; ++i) c[i] = 0;

    for (int ks = 0; ks < 8; ++ks) {
        v16bf a0 = *(const v16bf*)(wsA + (size_t)(mt0 * 8 + ks) * 512 + lane * 16);
        v16bf a1 = *(const v16bf*)(wsA + (size_t)(mt1 * 8 + ks) * 512 + lane * 16);
#pragma unroll
        for (int nt = 0; nt < 4; ++nt) {
            const __bf16* bp = Bs + (nt * 16 + nl) * BS_STRIDE + ks * 32 + kh * 8;
            v8bf blo = *(const v8bf*)(bp);
            v8bf bhi = *(const v8bf*)(bp + 16);
            v16bf bm = pack2(blo, bhi);
            c[2 * nt]     = wmma_bf(a0, bm, c[2 * nt]);
            c[2 * nt + 1] = wmma_bf(a1, bm, c[2 * nt + 1]);
        }
    }

#pragma unroll
    for (int nt = 0; nt < 4; ++nt) {
        int col = nbase + nt * 16 + nl;
        if (full || col < NC_) {
#pragma unroll
            for (int r = 0; r < 8; ++r) {
                int row0 = mt0 * 16 + r + (kh << 3);
                int row1 = mt1 * 16 + r + (kh << 3);
                out[(size_t)row0 * NC_ + col] = c[2 * nt][r];
                out[(size_t)row1 * NC_ + col] = c[2 * nt + 1][r];
            }
        }
    }
}

extern "C" void kernel_launch(void* const* d_in, const int* in_sizes, int n_in,
                              void* d_out, int out_size, void* d_ws, size_t ws_size,
                              hipStream_t stream) {
    const int*   inputs = (const int*)d_in[0];
    const float* masks  = (const float*)d_in[1];
    const float* emb    = (const float*)d_in[2];
    const float* Qw     = (const float*)d_in[3];
    const float* Kw     = (const float*)d_in[4];
    const float* lin_w  = (const float*)d_in[5];
    const float* lin_b  = (const float*)d_in[6];

    float* out_scores = (float*)d_out;                       // [B, N-1]
    float* out_aff    = out_scores + (size_t)B_ * NC_;       // [B, L, L]
    __bf16* wsA       = (__bf16*)d_ws;                       // swizzled final_rep, 512 KB

    sriem_session_kernel<<<dim3(B_), dim3(256), SMEM_TOTAL, stream>>>(
        inputs, masks, emb, Qw, Kw, lin_w, lin_b, out_aff, wsA);

    sriem_scores_kernel<<<dim3((NC_ + 63) / 64, 4), dim3(256), 0, stream>>>(
        wsA, emb, out_scores);
}
